// RGCN_25975962206900
// MI455X (gfx1250) — matrix operand order repaired and checked by reading, not compiled, not run on hardware
//
#include <hip/hip_runtime.h>
#include <hip/hip_bf16.h>

// ---------------------------------------------------------------------------
// RGCN on MI455X (gfx1250), basis-decomposed RelGraphConv.
//   x = relu(h @ W_elmo + b)  (concat zeros -> 512)
//   3x: Y_b = X @ V_b (bf16 WMMA, bf16 output -> per-layer hot set fits L2);
//       per-edge weighted gather + f32 atomic scatter; X = relu(ACC + b_gcn)
//   readout: partial per-graph sums -> mean -> logits -> softmax
// Weights are pre-transposed once so both GEMM operands stage with contiguous
// float4 loads + packed bf16 LDS stores. GEMM uses double-buffered LDS with
// one barrier per K-step.
// ---------------------------------------------------------------------------

#define TOT   16384            // BS*N
#define NEDGE 262144
#define HID   512
#define INDIM 3072
#define OUTD  128

typedef __attribute__((ext_vector_type(16))) __bf16 v16bf;
typedef __attribute__((ext_vector_type(8)))  __bf16 v8bf;
typedef __attribute__((ext_vector_type(4)))  __bf16 v4bf;
typedef __attribute__((ext_vector_type(8)))  float  v8f;

__device__ __forceinline__ v4bf cvt4(const float4 f) {
  v4bf r;
  r[0] = (__bf16)f.x; r[1] = (__bf16)f.y; r[2] = (__bf16)f.z; r[3] = (__bf16)f.w;
  return r;
}

__device__ __forceinline__ v16bf pack16(v8bf lo, v8bf hi) {
  v16bf r;
#pragma unroll
  for (int i = 0; i < 8; ++i) { r[i] = lo[i]; r[i + 8] = hi[i]; }
  return r;
}

// ---------------------------------------------------------------------------
// 32x32 tiled f32 transpose: BT[z][n][k] = B[z][k][n]. Run once per call on
// the (small) weight matrices.
// ---------------------------------------------------------------------------
__launch_bounds__(256)
__global__ void transpose_f32(const float* __restrict__ B, float* __restrict__ BT,
                              int K, int N, long zstride)
{
  __shared__ float t[32][33];
  const float* Bz  = B  + (long)blockIdx.z * zstride;
  float*       BTz = BT + (long)blockIdx.z * zstride;
  const int k0 = blockIdx.y * 32, n0 = blockIdx.x * 32;
  const int r = threadIdx.x >> 3;          // 0..31
  const int c = (threadIdx.x & 7) * 4;     // 0..28
  float4 v = *(const float4*)&Bz[(size_t)(k0 + r) * N + n0 + c];
  t[r][c + 0] = v.x; t[r][c + 1] = v.y; t[r][c + 2] = v.z; t[r][c + 3] = v.w;
  __syncthreads();
  float4 w;
  w.x = t[c + 0][r]; w.y = t[c + 1][r]; w.z = t[c + 2][r]; w.w = t[c + 3][r];
  *(float4*)&BTz[(size_t)(n0 + r) * K + k0 + c] = w;
}

// ---------------------------------------------------------------------------
// bf16 WMMA GEMM:  C[z] = act(A @ B[z] + bias),  B supplied pre-transposed
// ([n][k], row stride ldbt). Block: 256 threads (8 waves), tile 128x64, BK=32,
// double-buffered LDS (one barrier per K-step). Wave (wm,wn) owns 32x32 =
// 2x2 WMMA 16x16 tiles. Output f32 or bf16.
// ---------------------------------------------------------------------------
__launch_bounds__(256)
__global__ void gemm_bf16_wmma(const float* __restrict__ A, int lda,
                               const float* __restrict__ BT0, long btstride, int ldbt,
                               void* __restrict__ C0, long cstride, int ldc,
                               int K, const float* __restrict__ bias,
                               int do_relu, int out_bf16)
{
  const float* BT = BT0 + (long)blockIdx.z * btstride;
  const int m0 = blockIdx.y * 128;
  const int n0 = blockIdx.x * 64;
  const int tid  = threadIdx.x;
  const int lane = tid & 31;
  const int wm   = (tid >> 5) >> 1;   // 0..3
  const int wn   = (tid >> 5) & 1;    // 0..1

  __shared__ __bf16 sA[2][128 * 32];  // [m][k], stride 32
  __shared__ __bf16 sB[2][64 * 32];   // [n][k], stride 32

  v8f acc[2][2] = {};

  // staging indices: A = 16 floats per thread, B = 8 floats per thread
  const int arow  = tid >> 1;          // 0..127
  const int ahalf = (tid & 1) * 16;    // k: 0 | 16
  const int bnrow = tid >> 2;          // 0..63
  const int bseg  = (tid & 3) * 8;     // k: 0|8|16|24

  // fragment lane mapping (contiguous k = 16*(lane/16) + s, same for A and B; 16-bit A 16x32 / B 32x16, wave32)
  const int fr_m   = lane & 15;
  const int fr_klo = (lane < 16) ? 0 : 16;   // A: K 0-15 vs 16-31
  const int fr_kb  = (lane < 16) ? 0 : 16;   // B: K 0-15 vs 16-31

  const float* aptr = A  + (size_t)(m0 + arow)  * lda  + ahalf;
  const float* bptr = BT + (size_t)(n0 + bnrow) * ldbt + bseg;

  float4 a0, a1, a2, a3, b0, b1;

  auto stage = [&](int buf) {
    __bf16* da = &sA[buf][arow * 32 + ahalf];
    *(v4bf*)(da + 0)  = cvt4(a0);
    *(v4bf*)(da + 4)  = cvt4(a1);
    *(v4bf*)(da + 8)  = cvt4(a2);
    *(v4bf*)(da + 12) = cvt4(a3);
    __bf16* db = &sB[buf][bnrow * 32 + bseg];
    *(v4bf*)(db + 0) = cvt4(b0);
    *(v4bf*)(db + 4) = cvt4(b1);
  };

  auto compute = [&](int buf) {
    v16bf af[2], bfr[2];
#pragma unroll
    for (int i = 0; i < 2; ++i) {
      const __bf16* p = &sA[buf][(wm * 32 + i * 16 + fr_m) * 32];
      v8bf lo = *(const v8bf*)(p + fr_klo);
      v8bf hi = *(const v8bf*)(p + fr_klo + 8);
      af[i] = pack16(lo, hi);
    }
#pragma unroll
    for (int j = 0; j < 2; ++j)
      bfr[j] = *(const v16bf*)&sB[buf][(wn * 32 + j * 16 + fr_m) * 32 + fr_kb];
#pragma unroll
    for (int i = 0; i < 2; ++i)
#pragma unroll
      for (int j = 0; j < 2; ++j)
        acc[i][j] = __builtin_amdgcn_wmma_f32_16x16x32_bf16(
            false, af[i], false, bfr[j], (short)0, acc[i][j], false, false);
  };

  // prologue: tile 0
  a0 = ((const float4*)aptr)[0];
  a1 = ((const float4*)aptr)[1];
  a2 = ((const float4*)aptr)[2];
  a3 = ((const float4*)aptr)[3];
  b0 = ((const float4*)bptr)[0];
  b1 = ((const float4*)bptr)[1];
  stage(0);
  __syncthreads();

  const int nt = K >> 5;
  for (int t = 1; t < nt; ++t) {
    const float* ap = aptr + t * 32;
    const float* bp = bptr + t * 32;
    a0 = ((const float4*)ap)[0];
    a1 = ((const float4*)ap)[1];
    a2 = ((const float4*)ap)[2];
    a3 = ((const float4*)ap)[3];
    b0 = ((const float4*)bp)[0];
    b1 = ((const float4*)bp)[1];
    if (t + 1 < nt) __builtin_prefetch(aptr + (t + 1) * 32, 0, 1);  // global_prefetch_b8

    compute((t - 1) & 1);   // overlap WMMA with inflight global loads
    stage(t & 1);
    __syncthreads();
  }
  compute((nt - 1) & 1);

  // ---- epilogue: C/D layout (vgpr r -> M=r + 8*(lane>=16), N=lane&15) ----
#pragma unroll
  for (int i = 0; i < 2; ++i) {
#pragma unroll
    for (int j = 0; j < 2; ++j) {
      const int col = n0 + wn * 32 + j * 16 + (lane & 15);
      const int rw0 = m0 + wm * 32 + i * 16 + ((lane < 16) ? 0 : 8);
      const float ba = bias ? bias[col] : 0.0f;
#pragma unroll
      for (int r = 0; r < 8; ++r) {
        float v = acc[i][j][r] + ba;
        if (do_relu) v = fmaxf(v, 0.0f);
        if (out_bf16)
          ((__bf16*)C0 + (long)blockIdx.z * cstride)[(size_t)(rw0 + r) * ldc + col] = (__bf16)v;
        else
          ((float*)C0 + (long)blockIdx.z * cstride)[(size_t)(rw0 + r) * ldc + col] = v;
      }
    }
  }
}

// ---------------------------------------------------------------------------
// Edge phase: one wave per edge.
//   msg = norm_e * sum_b comp[r_e][b] * Y_b[src_e,:]  (Y in bf16, fits in L2)
//   -> f32 atomic add into ACC[dst_e]
// ---------------------------------------------------------------------------
__launch_bounds__(256)
__global__ void edge_scatter(const __bf16* __restrict__ Y,  // [4][TOT][HID] bf16
                             const int*   __restrict__ src,
                             const int*   __restrict__ dst,
                             const int*   __restrict__ rel,
                             const float* __restrict__ norm,
                             const float* __restrict__ comp, // [8][4]
                             float*       __restrict__ acc)  // [TOT][HID]
{
  const int lane = threadIdx.x & 31;
  const int e = blockIdx.x * 8 + (threadIdx.x >> 5);
  const int s  = src[e];
  const int d  = dst[e];
  const int rr = rel[e];
  const float w  = norm[e];
  const float c0 = w * comp[rr * 4 + 0];
  const float c1 = w * comp[rr * 4 + 1];
  const float c2 = w * comp[rr * 4 + 2];
  const float c3 = w * comp[rr * 4 + 3];

  const size_t zs = (size_t)TOT * HID;
  const __bf16* ys = Y + (size_t)s * HID;
  float* o = acc + (size_t)d * HID;

#pragma unroll
  for (int c = 0; c < 2; ++c) {
    const int col = c * 256 + lane * 8;
    v8bf y0 = *(const v8bf*)(ys + col);
    v8bf y1 = *(const v8bf*)(ys + zs + col);
    v8bf y2 = *(const v8bf*)(ys + 2 * zs + col);
    v8bf y3 = *(const v8bf*)(ys + 3 * zs + col);
#pragma unroll
    for (int j = 0; j < 8; ++j) {
      float v = c0 * (float)y0[j] + c1 * (float)y1[j]
              + c2 * (float)y2[j] + c3 * (float)y3[j];
      __hip_atomic_fetch_add(o + col + j, v, __ATOMIC_RELAXED, __HIP_MEMORY_SCOPE_AGENT);
    }
  }
}

// X = relu(ACC + b_gcn)  (float4 over TOT*HID)
__launch_bounds__(256)
__global__ void bias_relu(const float* __restrict__ acc,
                          const float* __restrict__ bgcn,
                          float* __restrict__ X)
{
  const int i = blockIdx.x * 256 + threadIdx.x;      // float4 index
  float4 a = ((const float4*)acc)[i];
  float4 b = ((const float4*)bgcn)[i & (HID / 4 - 1)];
  float4 v;
  v.x = fmaxf(a.x + b.x, 0.0f);
  v.y = fmaxf(a.y + b.y, 0.0f);
  v.z = fmaxf(a.z + b.z, 0.0f);
  v.w = fmaxf(a.w + b.w, 0.0f);
  ((float4*)X)[i] = v;
}

// Stage 1 of readout: P[g][slice][col] = sum over 64 nodes. grid (16,16).
__launch_bounds__(256)
__global__ void mean_partial(const float* __restrict__ X, float* __restrict__ P)
{
  const int sl = blockIdx.x;      // 0..15 (node slice)
  const int g  = blockIdx.y;      // 0..15 (graph)
  const float* Xg = X + ((size_t)g * 1024 + sl * 64) * HID;
  for (int col = threadIdx.x; col < HID; col += 256) {
    float s = 0.0f;
#pragma unroll 8
    for (int n = 0; n < 64; ++n) s += Xg[(size_t)n * HID + col];
    P[((size_t)g * 16 + sl) * HID + col] = s;
  }
}

// Stage 2: mean -> logits -> softmax. One block per graph.
__launch_bounds__(256)
__global__ void readout(const float* __restrict__ P,      // [16][16][HID]
                        const float* __restrict__ W_out,  // [HID][OUTD]
                        const float* __restrict__ b_out,  // [OUTD]
                        float* __restrict__ out)          // [16][OUTD]
{
  __shared__ float mean[HID];
  __shared__ float lg[OUTD];
  const int g   = blockIdx.x;
  const int tid = threadIdx.x;
  const float* Pg = P + (size_t)g * 16 * HID;

  for (int col = tid; col < HID; col += 256) {
    float s = 0.0f;
#pragma unroll
    for (int sl = 0; sl < 16; ++sl) s += Pg[sl * HID + col];
    mean[col] = s * (1.0f / 1024.0f);
  }
  __syncthreads();

  if (tid < OUTD) {
    float a = b_out[tid];
    for (int k = 0; k < HID; ++k) a += mean[k] * W_out[k * OUTD + tid];
    lg[tid] = a;
  }
  __syncthreads();

  if (tid == 0) {
    float mx = lg[0];
    for (int j = 1; j < OUTD; ++j) mx = fmaxf(mx, lg[j]);
    float s = 0.0f;
    for (int j = 0; j < OUTD; ++j) { float e = __expf(lg[j] - mx); lg[j] = e; s += e; }
    float inv = 1.0f / s;
    for (int j = 0; j < OUTD; ++j) out[g * OUTD + j] = lg[j] * inv;
  }
}

// ---------------------------------------------------------------------------
extern "C" void kernel_launch(void* const* d_in, const int* in_sizes, int n_in,
                              void* d_out, int out_size, void* d_ws, size_t ws_size,
                              hipStream_t stream)
{
  const float* h      = (const float*)d_in[0];
  const int*   src    = (const int*)d_in[1];
  const int*   dst    = (const int*)d_in[2];
  const int*   rel    = (const int*)d_in[3];
  const float* norm   = (const float*)d_in[4];
  const float* W_elmo = (const float*)d_in[5];
  const float* b_elmo = (const float*)d_in[6];
  const float* comp   = (const float*)d_in[7];
  const float* V      = (const float*)d_in[8];   // [4][512][512]
  const float* b_gcn  = (const float*)d_in[9];
  const float* W_out  = (const float*)d_in[10];
  const float* b_out  = (const float*)d_in[11];
  float* out = (float*)d_out;

  const size_t xbytes = (size_t)TOT * HID * sizeof(float);   // 32 MiB
  char* ws = (char*)d_ws;
  float*  X   = (float*)(ws);                                 // 32 MiB
  float*  ACC = (float*)(ws + xbytes);                        // 32 MiB
  __bf16* Y   = (__bf16*)(ws + 2 * xbytes);                   // 64 MiB (4 bases, bf16)
  float*  WT  = (float*)(ws + 2 * xbytes + 4 * (size_t)TOT * HID * sizeof(__bf16));
  float*  VT  = WT + (size_t)256 * INDIM;                     // 4 MiB
  float*  P   = VT + (size_t)4 * HID * HID;                   // 512 KiB

  // Pre-transpose weights: WT[256][3072], VT[4][512][512]
  transpose_f32<<<dim3(256 / 32, INDIM / 32, 1), 256, 0, stream>>>(
      W_elmo, WT, INDIM, 256, 0);
  transpose_f32<<<dim3(HID / 32, HID / 32, 4), 256, 0, stream>>>(
      V, VT, HID, HID, (long)HID * HID);

  // X = 0 provides the zero-concat upper 256 columns.
  hipMemsetAsync(X, 0, xbytes, stream);

  // ELMo: X[:, 0:256] = relu(h @ W_elmo + b_elmo)   (f32 output)
  gemm_bf16_wmma<<<dim3(256 / 64, TOT / 128, 1), 256, 0, stream>>>(
      h, INDIM, WT, 0, INDIM, X, 0, HID, INDIM, b_elmo, 1, 0);

  for (int layer = 0; layer < 3; ++layer) {
    hipMemsetAsync(ACC, 0, xbytes, stream);
    // Y_b = X @ V_b (bf16 output; per-layer hot set fits 192MB L2)
    gemm_bf16_wmma<<<dim3(HID / 64, TOT / 128, 4), 256, 0, stream>>>(
        X, HID, VT, (long)HID * HID, HID, Y, (long)TOT * HID, HID, HID, nullptr, 0, 1);
    edge_scatter<<<dim3(NEDGE / 8), 256, 0, stream>>>(Y, src, dst, rel, norm, comp, ACC);
    bias_relu<<<dim3((TOT * HID / 4) / 256), 256, 0, stream>>>(ACC, b_gcn, X);
  }

  mean_partial<<<dim3(16, 16), 256, 0, stream>>>(X, P);
  readout<<<dim3(16), 256, 0, stream>>>(P, W_out, b_out, out);
}
